// ReformerClassifier_73083163509421
// MI455X (gfx1250) — compile-verified, matrix operand order checked by reading, not run on hardware
//
#include <hip/hip_runtime.h>
#include <cstdint>
#include <cstddef>

// ---------------- model constants ----------------
#define TQ    8192      // T
#define DIMM  1024      // DIM
#define HH    8         // heads
#define DHH   128       // DIM/H
#define NHH   4         // hash rounds
#define NBB   128       // nb = T/BUCKET
#define NCC   512       // nc = NH*nb (chunks of 64)
#define BBATCH 2
#define FFD   4096
#define MTOK  16384     // B*T
#define NCLS  1000

// ---------------- WMMA types ----------------
typedef __attribute__((ext_vector_type(16))) __bf16 v16bf;
typedef __attribute__((ext_vector_type(8)))  float  v8f;

union FragBF { v16bf v; uint4 q[2]; };
union FragF  { v8f v; float f[8]; };

__device__ __forceinline__ unsigned short f2bf(float x) {
  union { float f; unsigned u; } c; c.f = x;
  unsigned r = c.u + 0x7FFFu + ((c.u >> 16) & 1u);
  return (unsigned short)(r >> 16);
}

// =====================================================================
// Embedding gather + pad mask;  x1 = x2 = emb[ids]
// =====================================================================
__global__ __launch_bounds__(256)
void k_embed(const int* __restrict__ ids, const float* __restrict__ emb,
             float* __restrict__ x1, float* __restrict__ x2, int* __restrict__ pad)
{
  int tok = blockIdx.x, tid = threadIdx.x;
  int id = ids[tok];
  if (tid == 0) pad[tok] = (id == 0) ? 1 : 0;
  const float* e = emb + (size_t)id * DIMM;
  float* o1 = x1 + (size_t)tok * DIMM;
  float* o2 = x2 + (size_t)tok * DIMM;
  for (int k = tid; k < DIMM; k += 256) { float v = e[k]; o1[k] = v; o2[k] = v; }
}

// =====================================================================
// One-time weight conversion: W[K][N] fp32 -> WT[N][K] bf16 (for async B staging)
// =====================================================================
__global__ __launch_bounds__(256)
void k_wconv(const float* __restrict__ src, unsigned short* __restrict__ dst,
             int K, int N)
{
  size_t idx = (size_t)blockIdx.x * 256 + threadIdx.x;
  if (idx >= (size_t)K * N) return;
  int k = (int)(idx / N), n = (int)(idx % N);
  dst[(size_t)n * K + k] = f2bf(src[idx]);
}

// =====================================================================
// LayerNorm over DIM=1024, one block per token
// =====================================================================
__global__ __launch_bounds__(256)
void k_ln(const float* __restrict__ x, const float* __restrict__ g,
          const float* __restrict__ bb, float* __restrict__ y)
{
  __shared__ float s1[256], s2[256];
  int tok = blockIdx.x, tid = threadIdx.x;
  const float* row = x + (size_t)tok * DIMM;
  float a = 0.f, b2 = 0.f;
  for (int e = tid; e < DIMM; e += 256) { float v = row[e]; a += v; b2 += v * v; }
  s1[tid] = a; s2[tid] = b2; __syncthreads();
  for (int s = 128; s > 0; s >>= 1) {
    if (tid < s) { s1[tid] += s1[tid + s]; s2[tid] += s2[tid + s]; }
    __syncthreads();
  }
  float mean = s1[0] * (1.f / DIMM);
  float var  = s2[0] * (1.f / DIMM) - mean * mean;
  float rstd = rsqrtf(var + 1e-5f);
  float* yr = y + (size_t)tok * DIMM;
  for (int e = tid; e < DIMM; e += 256) yr[e] = (row[e] - mean) * rstd * g[e] + bb[e];
}

// =====================================================================
// bf16 WMMA GEMM:  out = epilogue(A[MxK] @ W[KxN] + bias)
//   A: fp32 activations, converted to bf16 in-kernel (overlaps with async B)
//   W: pre-converted bf16, pre-transposed [N][K] -> B tile staged with
//      GLOBAL_LOAD_ASYNC_TO_LDS_B128 (ASYNCcnt), one 16B chunk per lane.
// MODE 1: GELU   2: head-split reshape   3: residual add
// block tile 128x64, BK=32, 8 wave32 waves, 32x32 per wave (2x2 WMMA)
// =====================================================================
#define GBM 128
#define GBN 64
#define GBK 32

template <int MODE>
__global__ __launch_bounds__(256)
void k_gemm(const float* __restrict__ A, const unsigned short* __restrict__ WT,
            const float* __restrict__ bias, float* __restrict__ out,
            const float* __restrict__ resid, int M, int N, int K)
{
  __shared__ __align__(16) unsigned short lA[GBM * GBK];
  __shared__ __align__(16) unsigned short lB[GBN * GBK]; // stored [n][k]

  const int tid = threadIdx.x;
  const int bm = blockIdx.y * GBM;
  const int bn = blockIdx.x * GBN;
  const int w = tid >> 5, lane = tid & 31, hf = lane >> 4, ln = lane & 15;
  const int wm = w >> 1, wn = w & 1;

  // per-thread async-B assignment: one b128 chunk (8 bf16) per lane per K-step
  const int brow = tid >> 2;          // 0..63  (n within tile)
  const int bchunk = (tid & 3) * 8;   // 0,8,16,24 (k within tile)
  const unsigned ldsB = (unsigned)(uintptr_t)(lB + brow * GBK + bchunk);

  FragF acc[2][2];
#pragma unroll
  for (int i = 0; i < 2; ++i)
#pragma unroll
    for (int j = 0; j < 2; ++j)
#pragma unroll
      for (int r = 0; r < 8; ++r) acc[i][j].f[r] = 0.f;

  for (int k0 = 0; k0 < K; k0 += GBK) {
    // --- issue async B tile copy first (overlaps with A load+convert) ---
    {
      const unsigned short* gp = WT + (size_t)(bn + brow) * K + k0 + bchunk;
      unsigned long long ga = (unsigned long long)(uintptr_t)gp;
      asm volatile("global_load_async_to_lds_b128 %0, %1, off"
                   :: "v"(ldsB), "v"(ga) : "memory");
    }
    // --- stage A (128x32 fp32 -> bf16) ---
#pragma unroll
    for (int it = 0; it < 4; ++it) {
      int q = tid + it * 256;
      int row = q >> 3, c4 = (q & 7) * 4;
      const float* ap = A + (size_t)(bm + row) * K + k0 + c4;
      if (it == 0 && k0 + GBK < K) __builtin_prefetch(ap + GBK, 0, 1);
      float4 fv = *(const float4*)ap;
      unsigned short* d = lA + row * GBK + c4;
      d[0] = f2bf(fv.x); d[1] = f2bf(fv.y); d[2] = f2bf(fv.z); d[3] = f2bf(fv.w);
    }
    asm volatile("s_wait_asynccnt 0" ::: "memory");
    __syncthreads();

    FragBF af[2], bfr[2];
#pragma unroll
    for (int mt = 0; mt < 2; ++mt) {
      const unsigned short* rp = lA + (wm * 32 + mt * 16 + ln) * GBK;
      af[mt].q[0] = *(const uint4*)(rp + hf * 8);
      af[mt].q[1] = *(const uint4*)(rp + 16 + hf * 8);
    }
#pragma unroll
    for (int nt = 0; nt < 2; ++nt) {
      const unsigned short* rp = lB + (wn * 32 + nt * 16 + ln) * GBK;
      bfr[nt].q[0] = *(const uint4*)(rp + hf * 8);
      bfr[nt].q[1] = *(const uint4*)(rp + 16 + hf * 8);
    }
#pragma unroll
    for (int mt = 0; mt < 2; ++mt)
#pragma unroll
      for (int nt = 0; nt < 2; ++nt)
        acc[mt][nt].v = __builtin_amdgcn_wmma_f32_16x16x32_bf16(
            false, af[mt].v, false, bfr[nt].v, (short)0, acc[mt][nt].v, false, false);
    __syncthreads();
  }

  // fused, branch-free (templated) epilogue
#pragma unroll
  for (int mt = 0; mt < 2; ++mt)
#pragma unroll
    for (int nt = 0; nt < 2; ++nt) {
      int n = bn + wn * 32 + nt * 16 + ln;
      float bi = bias ? bias[n] : 0.f;
#pragma unroll
      for (int r = 0; r < 8; ++r) {
        int m = bm + wm * 32 + mt * 16 + hf * 8 + r;
        float val = acc[mt][nt].f[r] + bi;
        if (MODE == 1) {
          val = 0.5f * val * (1.f + erff(val * 0.70710678118f));
          out[(size_t)m * N + n] = val;
        } else if (MODE == 2) {
          int b0 = m >> 13, t = m & (TQ - 1);
          int head = n >> 7, d = n & (DHH - 1);
          out[(((size_t)(b0 * HH + head)) * TQ + t) * DHH + d] = val;
        } else if (MODE == 3) {
          size_t idx = (size_t)m * N + n;
          out[idx] = resid[idx] + val;
        } else {
          out[(size_t)m * N + n] = val;
        }
      }
    }
}

// =====================================================================
// Deterministic pseudo-normal LSH rotations rot[d][h*64+i], 128x256
// =====================================================================
__device__ __forceinline__ unsigned hashu(unsigned x) {
  x ^= x >> 16; x *= 0x7feb352dU; x ^= x >> 15; x *= 0x846ca68bU; x ^= x >> 16;
  return x;
}
__global__ __launch_bounds__(256)
void k_rotgen(float* __restrict__ rot, int layer)
{
  int idx = blockIdx.x * 256 + threadIdx.x;
  if (idx >= 128 * 256) return;
  unsigned h1 = hashu((unsigned)idx * 2u + 1u + (unsigned)layer * 0x9E3779B9u);
  unsigned h2 = hashu((unsigned)idx * 2u + 2u + (unsigned)layer * 0x85EBCA77u);
  float u1 = (float)(h1 & 0xFFFFFFu) * (1.0f / 16777216.0f) + 1e-7f;
  float u2 = (float)(h2 & 0xFFFFFFu) * (1.0f / 16777216.0f);
  rot[idx] = sqrtf(-2.f * __logf(u1)) * __cosf(6.2831853f * u2);
}

// =====================================================================
// LSH bucketing: key[bh][h*T+t] = ((argmax([r,-r]) + h*nb) << 13) | t
// =====================================================================
__global__ __launch_bounds__(256)
void k_buckets(const float* __restrict__ qk, const float* __restrict__ rot,
               int* __restrict__ keys)
{
  __shared__ float row[128];
  __shared__ float sval[256];
  __shared__ int   sidx[256];
  int t = blockIdx.x, bh = blockIdx.y, tid = threadIdx.x;
  if (tid < 128) row[tid] = qk[((size_t)bh * TQ + t) * DHH + tid];
  __syncthreads();
  float a = 0.f;
  for (int d = 0; d < 128; ++d) a += row[d] * rot[d * 256 + tid];
  int i0 = tid & 63;
  float bv; int bi;
  if (a >= -a) { bv = a;  bi = i0; }
  else         { bv = -a; bi = 64 + i0; }
  sval[tid] = bv; sidx[tid] = bi; __syncthreads();
  for (int s = 32; s > 0; s >>= 1) {
    if ((tid & 63) < s) {
      float v2 = sval[tid + s]; int i2 = sidx[tid + s];
      if (v2 > sval[tid] || (v2 == sval[tid] && i2 < sidx[tid])) {
        sval[tid] = v2; sidx[tid] = i2;
      }
    }
    __syncthreads();
  }
  if ((tid & 63) == 0) {
    int h = tid >> 6;
    int bucket = sidx[tid] + h * NBB;
    keys[(size_t)bh * (NHH * TQ) + h * TQ + t] = (bucket << 13) | t;
  }
}

// =====================================================================
// Per-bh bitonic sort of 32768 unique keys in 128KB dynamic LDS
// =====================================================================
__global__ __launch_bounds__(1024)
void k_sort(const int* __restrict__ keys, int* __restrict__ perm)
{
  extern __shared__ char smemraw[];
  int* sk = (int*)smemraw;
  const int Nn = NHH * TQ;  // 32768
  int bh = blockIdx.x, tid = threadIdx.x;
  for (int i = tid; i < Nn; i += 1024) sk[i] = keys[(size_t)bh * Nn + i];
  __syncthreads();
  for (int k = 2; k <= Nn; k <<= 1) {
    for (int j = k >> 1; j > 0; j >>= 1) {
      for (int i = tid; i < Nn; i += 1024) {
        int l = i ^ j;
        if (l > i) {
          int a = sk[i], b = sk[l];
          bool up = ((i & k) == 0);
          if ((a > b) == up) { sk[i] = b; sk[l] = a; }
        }
      }
      __syncthreads();
    }
  }
  for (int i = tid; i < Nn; i += 1024) {
    int key = sk[i];
    int t = key & (TQ - 1);
    int bucket = key >> 13;
    int h = bucket >> 7;            // nb = 128
    perm[(size_t)bh * Nn + i] = h * TQ + t;
  }
}

// =====================================================================
// Fused LSH chunk attention (WMMA) — see round-1 comments.
// Dynamic LDS 83456B: sq(16K, later p) | sk(32K, later v^T) | dots(32K) | ids
// =====================================================================
__global__ __launch_bounds__(256)
void k_attn(const float* __restrict__ qk, const float* __restrict__ vv,
            const int* __restrict__ perm, const int* __restrict__ pad,
            float* __restrict__ orounds, float* __restrict__ lser)
{
  extern __shared__ char smemraw[];
  unsigned short* sq = (unsigned short*)(smemraw);           // 64x128 bf16 (q, then p)
  unsigned short* sk = (unsigned short*)(smemraw + 16384);   // 128x128 bf16 (k, then v^T)
  float* dots = (float*)(smemraw + 49152);                   // 64x128 f32
  int*   iq   = (int*)(smemraw + 81920);                     // 64
  int*   tk   = (int*)(smemraw + 82176);                     // 128
  int*   kp   = (int*)(smemraw + 82688);                     // 128
  float* lse  = (float*)(smemraw + 83200);                   // 64

  const int c = blockIdx.x, bh = blockIdx.y, tid = threadIdx.x;
  const int b0 = bh >> 3;
  const int* pb = perm + (size_t)bh * (NHH * TQ);

  if (tid < 64) {
    iq[tid] = pb[c * 64 + tid];
  } else if (tid < 192) {
    int j2 = tid - 64;
    int cc = (j2 < 64) ? c : ((c + NCC - 1) & (NCC - 1));
    int i = pb[cc * 64 + (j2 & 63)];
    int t = i & (TQ - 1);
    tk[j2] = t;
    kp[j2] = pad[b0 * TQ + t];
  }
  __syncthreads();

  {   // queries -> bf16 LDS (4 threads per row, 32 elems each)
    int row = tid >> 2, part = tid & 3;
    int t = iq[row] & (TQ - 1);
    const float* src = qk + ((size_t)bh * TQ + t) * DHH + part * 32;
    unsigned short* dst = sq + row * DHH + part * 32;
#pragma unroll
    for (int e = 0; e < 32; ++e) dst[e] = f2bf(src[e]);
  }
  if (tid < 128) {  // keys: l2-normalize then bf16
    const float* src = qk + ((size_t)bh * TQ + tk[tid]) * DHH;
    float ss = 0.f;
    for (int e = 0; e < 128; ++e) { float x = src[e]; ss += x * x; }
    float rn = rsqrtf(ss + 1e-12f);
    unsigned short* dst = sk + tid * DHH;
    for (int e = 0; e < 128; ++e) dst[e] = f2bf(src[e] * rn);
  }
  __syncthreads();

  const int w = tid >> 5, lane = tid & 31, hf = lane >> 4, ln = lane & 15;
  const int mi = w >> 1;          // m-tile 0..3 (M=64)
  const int ng = (w & 1) * 4;     // n-tile base 0 or 4 (N=128)

  // ---- phase 1: dots = bq @ bk^T (K=128) ----
  FragF acc[4];
#pragma unroll
  for (int nt = 0; nt < 4; ++nt)
#pragma unroll
    for (int r = 0; r < 8; ++r) acc[nt].f[r] = 0.f;

  for (int ks = 0; ks < 4; ++ks) {
    int k0 = ks * 32;
    FragBF af;
    {
      const unsigned short* rp = sq + (mi * 16 + ln) * DHH + k0;
      af.q[0] = *(const uint4*)(rp + hf * 8);
      af.q[1] = *(const uint4*)(rp + 16 + hf * 8);
    }
#pragma unroll
    for (int nt = 0; nt < 4; ++nt) {
      const unsigned short* rp = sk + ((ng + nt) * 16 + ln) * DHH + k0;
      FragBF bfr;
      bfr.q[0] = *(const uint4*)(rp + hf * 8);
      bfr.q[1] = *(const uint4*)(rp + 16 + hf * 8);
      acc[nt].v = __builtin_amdgcn_wmma_f32_16x16x32_bf16(
          false, af.v, false, bfr.v, (short)0, acc[nt].v, false, false);
    }
  }
  const float scale = 0.08838834764831845f;   // DH^-0.5
#pragma unroll
  for (int nt = 0; nt < 4; ++nt) {
    int ncol = (ng + nt) * 16 + ln;
    int tkey = tk[ncol];
    int kpd = kp[ncol];
#pragma unroll
    for (int r = 0; r < 8; ++r) {
      int m = mi * 16 + hf * 8 + r;
      float val = acc[nt].f[r] * scale;
      if (kpd) val = -1e9f;                        // pad mask
      if ((iq[m] & (TQ - 1)) == tkey) val = -5e4f; // shared-QK self mask
      dots[m * 128 + ncol] = val;
    }
  }
  __syncthreads();

  // ---- phase 2: load v^T (overwrites sk) + softmax (p overwrites sq) ----
  if (tid < 128) {
    const float* src = vv + ((size_t)bh * TQ + tk[tid]) * DHH;
    for (int e = 0; e < 128; ++e) sk[e * 128 + tid] = f2bf(src[e]); // svT[d][j2]
  } else if (tid < 192) {
    int row = tid - 128;
    float mx = -1e30f;
    for (int e = 0; e < 128; ++e) mx = fmaxf(mx, dots[row * 128 + e]);
    float s = 0.f;
    for (int e = 0; e < 128; ++e) s += __expf(dots[row * 128 + e] - mx);
    float l = mx + __logf(s);
    lse[row] = l;
    for (int e = 0; e < 128; ++e) sq[row * 128 + e] = f2bf(__expf(dots[row * 128 + e] - l));
    int i = iq[row];
    int h = i >> 13, t = i & (TQ - 1);
    lser[((size_t)bh * NHH + h) * TQ + t] = l;
  }
  __syncthreads();

  // ---- phase 3: bo = p @ bvv (K=128 over keys) ----
  FragF oacc[4];
#pragma unroll
  for (int nt = 0; nt < 4; ++nt)
#pragma unroll
    for (int r = 0; r < 8; ++r) oacc[nt].f[r] = 0.f;

  for (int ks = 0; ks < 4; ++ks) {
    int k0 = ks * 32;
    FragBF af;
    {
      const unsigned short* rp = sq + (mi * 16 + ln) * DHH + k0;  // p
      af.q[0] = *(const uint4*)(rp + hf * 8);
      af.q[1] = *(const uint4*)(rp + 16 + hf * 8);
    }
#pragma unroll
    for (int nt = 0; nt < 4; ++nt) {
      const unsigned short* rp = sk + ((ng + nt) * 16 + ln) * 128 + k0; // svT[d][j2]
      FragBF bfr;
      bfr.q[0] = *(const uint4*)(rp + hf * 8);
      bfr.q[1] = *(const uint4*)(rp + 16 + hf * 8);
      oacc[nt].v = __builtin_amdgcn_wmma_f32_16x16x32_bf16(
          false, af.v, false, bfr.v, (short)0, oacc[nt].v, false, false);
    }
  }
  // scatter back through perm
#pragma unroll
  for (int nt = 0; nt < 4; ++nt) {
    int d = (ng + nt) * 16 + ln;
#pragma unroll
    for (int r = 0; r < 8; ++r) {
      int m = mi * 16 + hf * 8 + r;
      int i = iq[m];
      int h = i >> 13, t = i & (TQ - 1);
      orounds[(((size_t)bh * NHH + h) * TQ + t) * DHH + d] = oacc[nt].f[r];
    }
  }
}

// =====================================================================
// Combine hash rounds: softmax over per-round LSE, write merged heads
// =====================================================================
__global__ __launch_bounds__(128)
void k_combine(const float* __restrict__ orounds, const float* __restrict__ lser,
               float* __restrict__ attn)
{
  int t = blockIdx.x, bh = blockIdx.y, d = threadIdx.x;
  float l[4];
#pragma unroll
  for (int h = 0; h < 4; ++h) l[h] = lser[((size_t)bh * NHH + h) * TQ + t];
  float mx = fmaxf(fmaxf(l[0], l[1]), fmaxf(l[2], l[3]));
  float e[4], s = 0.f;
#pragma unroll
  for (int h = 0; h < 4; ++h) { e[h] = __expf(l[h] - mx); s += e[h]; }
  float inv = 1.f / s;
  float o = 0.f;
#pragma unroll
  for (int h = 0; h < 4; ++h)
    o += (e[h] * inv) * orounds[(((size_t)bh * NHH + h) * TQ + t) * DHH + d];
  int b0 = bh >> 3, head = bh & 7;
  attn[((size_t)(b0 * TQ + t)) * DIMM + head * DHH + d] = o;
}

// =====================================================================
// CLS pooling + classifier head (tiny M=2 GEMMs)
// =====================================================================
__global__ __launch_bounds__(128)
void k_pooled(const float* __restrict__ x1, const float* __restrict__ x2,
              const float* __restrict__ Wpre, const float* __restrict__ bpre,
              float* __restrict__ pooled)
{
  int n = blockIdx.x * 128 + threadIdx.x;
  int b = blockIdx.y;
  const float* r1 = x1 + (size_t)b * TQ * DIMM;
  const float* r2 = x2 + (size_t)b * TQ * DIMM;
  float a = 0.f;
  for (int k = 0; k < DIMM; ++k) {
    float h = 0.5f * (r1[k] + r2[k]);
    a += h * Wpre[(size_t)k * DIMM + n];
  }
  pooled[b * DIMM + n] = tanhf(a + bpre[n]);
}

__global__ __launch_bounds__(128)
void k_logits(const float* __restrict__ pooled, const float* __restrict__ Wcls,
              const float* __restrict__ bcls, float* __restrict__ out)
{
  int n = blockIdx.x * 128 + threadIdx.x;
  int b = blockIdx.y;
  if (n >= NCLS) return;
  float a = 0.f;
  for (int k = 0; k < DIMM; ++k) a += pooled[b * DIMM + k] * Wcls[(size_t)k * NCLS + n];
  out[b * NCLS + n] = a + bcls[n];
}

// =====================================================================
// Workspace layout (float units)
// =====================================================================
static const size_t SZ_ACT  = (size_t)MTOK * DIMM;                 // 16,777,216
static const size_t SZ_BIG  = (size_t)16 * NHH * TQ * DHH;         // 67,108,864
static const size_t OFF_X1   = 0;
static const size_t OFF_X2   = OFF_X1 + SZ_ACT;
static const size_t OFF_TLN  = OFF_X2 + SZ_ACT;
static const size_t OFF_QK   = OFF_TLN + SZ_ACT;
static const size_t OFF_V    = OFF_QK + SZ_ACT;
static const size_t OFF_ATTN = OFF_V + SZ_ACT;
static const size_t OFF_BIG  = OFF_ATTN + SZ_ACT;                  // ff1 / o_rounds alias
static const size_t OFF_LSE  = OFF_BIG + SZ_BIG;                   // 524,288
static const size_t OFF_ROT  = OFF_LSE + (size_t)16 * NHH * TQ;    // 32,768
static const size_t OFF_POOL = OFF_ROT + 128 * 256;                // 2,048
static const size_t OFF_KEYS = OFF_POOL + BBATCH * DIMM;           // int 524,288
static const size_t OFF_PERM = OFF_KEYS + (size_t)16 * NHH * TQ;   // int 524,288
static const size_t OFF_PAD  = OFF_PERM + (size_t)16 * NHH * TQ;   // int 16,384
static const size_t OFF_WBF  = OFF_PAD + (size_t)MTOK;             // bf16 weights: 23,068,672 ushort

// bf16 transposed weight offsets (ushort units, within WB)
static const size_t WB_QK = 0;              // 2 x 1M, layer stride 1048576
static const size_t WB_V  = 2097152;
static const size_t WB_O  = 4194304;
static const size_t WB_1  = 6291456;        // 2 x 4M, layer stride 4194304
static const size_t WB_2  = 14680064;

extern "C" void kernel_launch(void* const* d_in, const int* in_sizes, int n_in,
                              void* d_out, int out_size, void* d_ws, size_t ws_size,
                              hipStream_t stream)
{
  (void)in_sizes; (void)n_in; (void)out_size; (void)ws_size;

  const int*   xids = (const int*)d_in[0];
  // d_in[1] = source_lengths (unused by the reference forward)
  const float* emb  = (const float*)d_in[2];
  const float* ln1g = (const float*)d_in[3];
  const float* ln1b = (const float*)d_in[4];
  const float* Wqk  = (const float*)d_in[5];
  const float* Wv   = (const float*)d_in[6];
  const float* Wo   = (const float*)d_in[7];
  const float* bo   = (const float*)d_in[8];
  const float* ln2g = (const float*)d_in[9];
  const float* ln2b = (const float*)d_in[10];
  const float* W1   = (const float*)d_in[11];
  const float* b1   = (const float*)d_in[12];
  const float* W2   = (const float*)d_in[13];
  const float* b2   = (const float*)d_in[14];
  const float* Wpre = (const float*)d_in[15];
  const float* bpre = (const float*)d_in[16];
  const float* Wcls = (const float*)d_in[17];
  const float* bcls = (const float*)d_in[18];
  float* out = (float*)d_out;

  float* ws   = (float*)d_ws;
  float* x1   = ws + OFF_X1;
  float* x2   = ws + OFF_X2;
  float* tln  = ws + OFF_TLN;
  float* qkb  = ws + OFF_QK;
  float* vb   = ws + OFF_V;
  float* attn = ws + OFF_ATTN;
  float* big  = ws + OFF_BIG;      // ff1 and o_rounds (lifetime-disjoint)
  float* lse  = ws + OFF_LSE;
  float* rot  = ws + OFF_ROT;
  float* pool = ws + OFF_POOL;
  int*   keys = (int*)(ws + OFF_KEYS);
  int*   perm = (int*)(ws + OFF_PERM);
  int*   pad  = (int*)(ws + OFF_PAD);
  unsigned short* WB = (unsigned short*)(ws + OFF_WBF);

  // one-time (per launch) bf16 + transpose weight conversion
  for (int L = 0; L < 2; ++L) {
    k_wconv<<<4096, 256, 0, stream>>>(Wqk + (size_t)L * DIMM * DIMM,
                                      WB + WB_QK + (size_t)L * 1048576, DIMM, DIMM);
    k_wconv<<<4096, 256, 0, stream>>>(Wv + (size_t)L * DIMM * DIMM,
                                      WB + WB_V + (size_t)L * 1048576, DIMM, DIMM);
    k_wconv<<<4096, 256, 0, stream>>>(Wo + (size_t)L * DIMM * DIMM,
                                      WB + WB_O + (size_t)L * 1048576, DIMM, DIMM);
    k_wconv<<<16384, 256, 0, stream>>>(W1 + (size_t)L * DIMM * FFD,
                                       WB + WB_1 + (size_t)L * 4194304, DIMM, FFD);
    k_wconv<<<16384, 256, 0, stream>>>(W2 + (size_t)L * FFD * DIMM,
                                       WB + WB_2 + (size_t)L * 4194304, FFD, DIMM);
  }

  k_embed<<<MTOK, 256, 0, stream>>>(xids, emb, x1, x2, pad);

  for (int L = 0; L < 2; ++L) {
    // --- attention sub-block ---
    k_ln<<<MTOK, 256, 0, stream>>>(x2, ln1g + L * DIMM, ln1b + L * DIMM, tln);
    k_gemm<2><<<dim3(DIMM / GBN, MTOK / GBM), 256, 0, stream>>>(
        tln, WB + WB_QK + (size_t)L * 1048576, nullptr, qkb, nullptr, MTOK, DIMM, DIMM);
    k_gemm<2><<<dim3(DIMM / GBN, MTOK / GBM), 256, 0, stream>>>(
        tln, WB + WB_V + (size_t)L * 1048576, nullptr, vb, nullptr, MTOK, DIMM, DIMM);

    k_rotgen<<<128, 256, 0, stream>>>(rot, L);
    k_buckets<<<dim3(TQ, 16), 256, 0, stream>>>(qkb, rot, keys);
    k_sort<<<16, 1024, (size_t)(NHH * TQ) * sizeof(int), stream>>>(keys, perm);
    k_attn<<<dim3(NCC, 16), 256, 83456, stream>>>(qkb, vb, perm, pad, big, lse);
    k_combine<<<dim3(TQ, 16), 128, 0, stream>>>(big, lse, attn);

    k_gemm<3><<<dim3(DIMM / GBN, MTOK / GBM), 256, 0, stream>>>(
        attn, WB + WB_O + (size_t)L * 1048576, bo + L * DIMM, x1, x1, MTOK, DIMM, DIMM);

    // --- feed-forward sub-block ---
    k_ln<<<MTOK, 256, 0, stream>>>(x1, ln2g + L * DIMM, ln2b + L * DIMM, tln);
    k_gemm<1><<<dim3(FFD / GBN, MTOK / GBM), 256, 0, stream>>>(
        tln, WB + WB_1 + (size_t)L * 4194304, b1 + L * FFD, big, nullptr, MTOK, FFD, DIMM);
    k_gemm<3><<<dim3(DIMM / GBN, MTOK / GBM), 256, 0, stream>>>(
        big, WB + WB_2 + (size_t)L * 4194304, b2 + L * DIMM, x2, x2, MTOK, DIMM, FFD);
  }

  k_pooled<<<dim3(DIMM / 128, BBATCH), 128, 0, stream>>>(x1, x2, Wpre, bpre, pool);
  k_logits<<<dim3(8, BBATCH), 128, 0, stream>>>(pool, Wcls, bcls, out);
}
